// SFOLayer_56427280335283
// MI455X (gfx1250) — matrix operand order repaired and checked by reading, not compiled, not run on hardware
//
#include <hip/hip_runtime.h>
#include <cstdint>

// ---------------------------------------------------------------------------
// SFO / mini-STU layer for MI455X (gfx1250), wave32 + WMMA bf16.
// Strategy: causal FFT conv == lower-triangular Toeplitz GEMM banks.
// All heavy math runs through v_wmma_f32_16x16x32_bf16.
// ---------------------------------------------------------------------------

typedef __attribute__((ext_vector_type(16))) __bf16 v16bf;
typedef __attribute__((ext_vector_type(8)))  float  v8f;
typedef __attribute__((ext_vector_type(8)))  unsigned int v8u;

#define SEQ_L   128
#define DIM     64
#define NFILT   16
#define HID_DIM 256

union FragU { v8u u; v16bf h; };

// A-operand fragment (16x32 bf16): lane m = lane&15, K interleave per ISA table:
// lanes 0-15: V0-3 -> K 0..7, V4-7 -> K 16..23 ; lanes 16-31: K 8..15 / 24..31.
__device__ __forceinline__ v16bf load_fragA(const __bf16* base, int stride,
                                            int row, int kbase) {
  const uint4* p0 = (const uint4*)(base + row * stride + kbase);
  const uint4* p1 = (const uint4*)(base + row * stride + kbase + 16);
  uint4 a = *p0; uint4 b = *p1;
  FragU f;
  f.u[0] = a.x; f.u[1] = a.y; f.u[2] = a.z; f.u[3] = a.w;
  f.u[4] = b.x; f.u[5] = b.y; f.u[6] = b.z; f.u[7] = b.w;
  return f.h;
}

// B-operand fragment (32x16 bf16) from TRANSPOSED storage BT[n][k]:
// lane n = lane&15; lanes 0-15 hold K 0..7 / 16..23, lanes 16-31 hold K 8..15 / 24..31
// (same interleave as A -> two b128 loads 16 K apart).
__device__ __forceinline__ v16bf load_fragB(const __bf16* base, int stride,
                                            int row, int kbase16) {
  const uint4* p0 = (const uint4*)(base + row * stride + kbase16);
  const uint4* p1 = (const uint4*)(base + row * stride + kbase16 + 16);
  uint4 a = *p0; uint4 b = *p1;
  FragU f;
  f.u[0] = a.x; f.u[1] = a.y; f.u[2] = a.z; f.u[3] = a.w;
  f.u[4] = b.x; f.u[5] = b.y; f.u[6] = b.z; f.u[7] = b.w;
  return f.h;
}

__device__ __forceinline__ v8f wmma_bf16(v16bf a, v16bf b, v8f c) {
  // (neg_a, A, neg_b, B, c_mod, C, reuse_a, reuse_b)
  return __builtin_amdgcn_wmma_f32_16x16x32_bf16(false, a, false, b,
                                                 (short)0, c, false, false);
}

// Store a 16x16 f32 C/D tile (rows across VGPRs, cols across lanes) as bf16
// into a row-major LDS buffer.
__device__ __forceinline__ void store_tile(__bf16* buf, int stride, int colbase,
                                           v8f acc, int lane) {
  const int g = lane >> 4;
  const int n = lane & 15;
#pragma unroll
  for (int v = 0; v < 8; ++v)
    buf[(v + 8 * g) * stride + colbase + n] = (__bf16)acc[v];
}

__device__ __forceinline__ float gelu_f(float x) {
  float x3 = x * x * x;
  return 0.5f * x * (1.0f + tanhf(0.79788456080286536f * (x + 0.044715f * x3)));
}

// ---------------------------------------------------------------------------
// Precompute: bf16 Toeplitz banks Tp[k][l][s] = phi[l-s,k] (causal),
// Tm[k][l][s] = (-1)^(l-s) * phi[l-s,k].
// ---------------------------------------------------------------------------
__global__ void build_toeplitz_kernel(const float* __restrict__ phi,
                                      __bf16* __restrict__ Tp,
                                      __bf16* __restrict__ Tm) {
  int idx = blockIdx.x * blockDim.x + threadIdx.x;
  if (idx >= NFILT * SEQ_L * SEQ_L) return;
  int s = idx & 127;
  int l = (idx >> 7) & 127;
  int k = idx >> 14;
  float vp = 0.0f, vm = 0.0f;
  int t = l - s;
  if (t >= 0) {
    float p = phi[t * NFILT + k];
    vp = p;
    vm = (t & 1) ? -p : p;
  }
  Tp[idx] = (__bf16)vp;
  Tm[idx] = (__bf16)vm;
}

// Transpose each (R,C) slice of an f32 tensor into (C,R) bf16.
__global__ void transpose_cast_kernel(const float* __restrict__ src,
                                      __bf16* __restrict__ dst,
                                      int slices, int R, int C) {
  int idx = blockIdx.x * blockDim.x + threadIdx.x;
  int total = slices * R * C;
  if (idx >= total) return;
  int c = idx % C;
  int r = (idx / C) % R;
  int sl = idx / (R * C);
  dst[(size_t)sl * R * C + (size_t)c * R + r] = (__bf16)src[idx];
}

// ---------------------------------------------------------------------------
// Main pass kernel. PASS=0: row pass (sequence along W); PASS=1: col pass.
// One workgroup (8 wave32) per sequence; wave w owns output rows [16w,16w+16).
// ---------------------------------------------------------------------------
template <int PASS>
__global__ __launch_bounds__(256, 1) void stu_pass_kernel(
    const float* __restrict__ xin, float* __restrict__ xout,
    const float* __restrict__ gamma, const float* __restrict__ beta,
    const __bf16* __restrict__ Tp, const __bf16* __restrict__ Tm,
    const __bf16* __restrict__ mpT, const __bf16* __restrict__ mmT,
    const __bf16* __restrict__ w1T, const float* __restrict__ b1,
    const __bf16* __restrict__ w2T, const float* __restrict__ b2) {
  constexpr int XSTR = 136;  // padded stride (16B aligned rows, conflict relief)
  __shared__ __attribute__((aligned(16))) __bf16 xnT[DIM * XSTR];   // [d][s]
  __shared__ __attribute__((aligned(16))) __bf16 stage[8][2][16 * DIM];

  const int seq = blockIdx.x;     // 0..511
  const int bidx = seq >> 7;      // batch
  const int rc = seq & 127;       // row (PASS 0) / col (PASS 1) index
  const int tid = threadIdx.x;
  const int lane = tid & 31;
  const int wav = tid >> 5;

  auto gidx = [&](int l, int ch) -> size_t {
    if (PASS == 0)
      return (((size_t)bidx * 128 + rc) * 128 + (size_t)l) * 64 + ch;
    else
      return (((size_t)bidx * 128 + (size_t)l) * 128 + rc) * 64 + ch;
  };

  // ---- LayerNorm: threads 0..127 each own one sequence position ----
  if (tid < 128) {
    const float* rp = xin + gidx(tid, 0);
    float vals[64];
    float mu = 0.0f;
#pragma unroll
    for (int i = 0; i < 64; ++i) { vals[i] = rp[i]; mu += vals[i]; }
    mu *= (1.0f / 64.0f);
    float var = 0.0f;
#pragma unroll
    for (int i = 0; i < 64; ++i) { float d = vals[i] - mu; var += d * d; }
    var *= (1.0f / 64.0f);
    float rs = rsqrtf(var + 1e-5f);
#pragma unroll
    for (int i = 0; i < 64; ++i)
      xnT[i * XSTR + tid] = (__bf16)((vals[i] - mu) * rs * gamma[i] + beta[i]);
  }
  __syncthreads();

  const int l0 = wav * 16;
  const int g = lane >> 4;
  const int g8 = g * 8;
  const int g16 = g * 16;
  const int m16 = lane & 15;

  const v8f zf = {0.f, 0.f, 0.f, 0.f, 0.f, 0.f, 0.f, 0.f};
  v8f spec[4] = {zf, zf, zf, zf};

  __bf16* ub = &stage[wav][0][0];
  __bf16* mb = &stage[wav][1][0];

  // ---- K filters: causal Toeplitz conv + per-filter projection ----
#pragma unroll 1
  for (int k = 0; k < NFILT; ++k) {
    const __bf16* TpK = Tp + k * SEQ_L * SEQ_L;
    const __bf16* TmK = Tm + k * SEQ_L * SEQ_L;
    v8f up[4] = {zf, zf, zf, zf};
    v8f um[4] = {zf, zf, zf, zf};
#pragma unroll
    for (int kk = 0; kk < 4; ++kk) {
      v16bf ap = load_fragA(TpK, SEQ_L, l0 + m16, kk * 32 + g8);
      v16bf am = load_fragA(TmK, SEQ_L, l0 + m16, kk * 32 + g8);
#pragma unroll
      for (int nt = 0; nt < 4; ++nt) {
        v16bf bx = load_fragB(xnT, XSTR, nt * 16 + m16, kk * 32 + g8);
        up[nt] = wmma_bf16(ap, bx, up[nt]);
        um[nt] = wmma_bf16(am, bx, um[nt]);
      }
    }
    // stage Up/Um tiles (f32 -> bf16) for use as A operands of projection
#pragma unroll
    for (int nt = 0; nt < 4; ++nt) {
      store_tile(ub, DIM, nt * 16, up[nt], lane);
      store_tile(mb, DIM, nt * 16, um[nt], lane);
    }
    asm volatile("s_wait_dscnt 0" ::: "memory");

    const __bf16* mpK = mpT + k * DIM * DIM;  // transposed: [e][d]
    const __bf16* mmK = mmT + k * DIM * DIM;
#pragma unroll
    for (int j = 0; j < 2; ++j) {
      v16bf aup = load_fragA(ub, DIM, m16, j * 32 + g8);
      v16bf aum = load_fragA(mb, DIM, m16, j * 32 + g8);
#pragma unroll
      for (int nt = 0; nt < 4; ++nt) {
        v16bf bp = load_fragB(mpK, DIM, nt * 16 + m16, j * 32 + g8);
        v16bf bm = load_fragB(mmK, DIM, nt * 16 + m16, j * 32 + g8);
        spec[nt] = wmma_bf16(aup, bp, spec[nt]);
        spec[nt] = wmma_bf16(aum, bm, spec[nt]);
      }
    }
  }

  // ---- MLP: out = gelu(spec @ w1 + b1) @ w2 + b2 (HID in 4 chunks of 64) ----
#pragma unroll
  for (int nt = 0; nt < 4; ++nt) store_tile(ub, DIM, nt * 16, spec[nt], lane);
  asm volatile("s_wait_dscnt 0" ::: "memory");

  v8f outa[4] = {zf, zf, zf, zf};
#pragma unroll 1
  for (int c = 0; c < 4; ++c) {
    v8f h[4] = {zf, zf, zf, zf};
#pragma unroll
    for (int j = 0; j < 2; ++j) {
      v16bf as = load_fragA(ub, DIM, m16, j * 32 + g8);
#pragma unroll
      for (int nt = 0; nt < 4; ++nt) {
        v16bf bw = load_fragB(w1T, DIM, c * 64 + nt * 16 + m16, j * 32 + g8);
        h[nt] = wmma_bf16(as, bw, h[nt]);
      }
    }
    // bias + GELU, stage as bf16 A-tile
#pragma unroll
    for (int nt = 0; nt < 4; ++nt) {
      float bias = b1[c * 64 + nt * 16 + m16];
#pragma unroll
      for (int v = 0; v < 8; ++v)
        mb[(v + g8) * DIM + nt * 16 + m16] = (__bf16)gelu_f(h[nt][v] + bias);
    }
    asm volatile("s_wait_dscnt 0" ::: "memory");
#pragma unroll
    for (int j = 0; j < 2; ++j) {
      v16bf ah = load_fragA(mb, DIM, m16, j * 32 + g8);
#pragma unroll
      for (int nt = 0; nt < 4; ++nt) {
        v16bf bw2 = load_fragB(w2T, HID_DIM, nt * 16 + m16,
                               c * 64 + j * 32 + g8);
        outa[nt] = wmma_bf16(ah, bw2, outa[nt]);
      }
    }
  }

  // ---- epilogue: +b2, residual with original x, store ----
#pragma unroll
  for (int nt = 0; nt < 4; ++nt) {
    float bias = b2[nt * 16 + m16];
#pragma unroll
    for (int v = 0; v < 8; ++v) {
      int row = l0 + v + g8;
      size_t a = gidx(row, nt * 16 + m16);
      xout[a] = outa[nt][v] + bias + xin[a];
    }
  }
}

// ---------------------------------------------------------------------------
// Host launcher
// ---------------------------------------------------------------------------
extern "C" void kernel_launch(void* const* d_in, const int* in_sizes, int n_in,
                              void* d_out, int out_size, void* d_ws,
                              size_t ws_size, hipStream_t stream) {
  (void)in_sizes; (void)n_in; (void)out_size; (void)ws_size;

  const float* v         = (const float*)d_in[0];
  const float* gamma_row = (const float*)d_in[1];
  const float* beta_row  = (const float*)d_in[2];
  const float* gamma_col = (const float*)d_in[3];
  const float* beta_col  = (const float*)d_in[4];
  const float* mp_row    = (const float*)d_in[5];
  const float* mm_row    = (const float*)d_in[6];
  const float* mp_col    = (const float*)d_in[7];
  const float* mm_col    = (const float*)d_in[8];
  const float* w1_row    = (const float*)d_in[9];
  const float* b1_row    = (const float*)d_in[10];
  const float* w2_row    = (const float*)d_in[11];
  const float* b2_row    = (const float*)d_in[12];
  const float* w1_col    = (const float*)d_in[13];
  const float* b1_col    = (const float*)d_in[14];
  const float* w2_col    = (const float*)d_in[15];
  const float* b2_col    = (const float*)d_in[16];
  const float* phi_row   = (const float*)d_in[17];
  const float* phi_col   = (const float*)d_in[18];

  __bf16* ws = (__bf16*)d_ws;
  const size_t TSZ = (size_t)NFILT * SEQ_L * SEQ_L;  // 262144
  const size_t MSZ = (size_t)NFILT * DIM * DIM;      // 65536
  const size_t WSZ = (size_t)DIM * HID_DIM;          // 16384

  __bf16* Tp_row  = ws;
  __bf16* Tm_row  = Tp_row + TSZ;
  __bf16* Tp_col  = Tm_row + TSZ;
  __bf16* Tm_col  = Tp_col + TSZ;
  __bf16* mpT_row = Tm_col + TSZ;
  __bf16* mmT_row = mpT_row + MSZ;
  __bf16* mpT_col = mmT_row + MSZ;
  __bf16* mmT_col = mpT_col + MSZ;
  __bf16* w1T_row = mmT_col + MSZ;
  __bf16* w2T_row = w1T_row + WSZ;
  __bf16* w1T_col = w2T_row + WSZ;
  __bf16* w2T_col = w1T_col + WSZ;
  __bf16* end_bf  = w2T_col + WSZ;
  float* xmid = (float*)(((uintptr_t)end_bf + 255) & ~(uintptr_t)255);

  const int THR = 256;
  build_toeplitz_kernel<<<(int)((TSZ + THR - 1) / THR), THR, 0, stream>>>(
      phi_row, Tp_row, Tm_row);
  build_toeplitz_kernel<<<(int)((TSZ + THR - 1) / THR), THR, 0, stream>>>(
      phi_col, Tp_col, Tm_col);
  transpose_cast_kernel<<<(int)((MSZ + THR - 1) / THR), THR, 0, stream>>>(
      mp_row, mpT_row, NFILT, DIM, DIM);
  transpose_cast_kernel<<<(int)((MSZ + THR - 1) / THR), THR, 0, stream>>>(
      mm_row, mmT_row, NFILT, DIM, DIM);
  transpose_cast_kernel<<<(int)((MSZ + THR - 1) / THR), THR, 0, stream>>>(
      mp_col, mpT_col, NFILT, DIM, DIM);
  transpose_cast_kernel<<<(int)((MSZ + THR - 1) / THR), THR, 0, stream>>>(
      mm_col, mmT_col, NFILT, DIM, DIM);
  transpose_cast_kernel<<<(int)((WSZ + THR - 1) / THR), THR, 0, stream>>>(
      w1_row, w1T_row, 1, DIM, HID_DIM);
  transpose_cast_kernel<<<(int)((WSZ + THR - 1) / THR), THR, 0, stream>>>(
      w2_row, w2T_row, 1, HID_DIM, DIM);
  transpose_cast_kernel<<<(int)((WSZ + THR - 1) / THR), THR, 0, stream>>>(
      w1_col, w1T_col, 1, DIM, HID_DIM);
  transpose_cast_kernel<<<(int)((WSZ + THR - 1) / THR), THR, 0, stream>>>(
      w2_col, w2T_col, 1, HID_DIM, DIM);

  // Row pass: v -> xmid (with residual)
  stu_pass_kernel<0><<<512, 256, 0, stream>>>(
      v, xmid, gamma_row, beta_row, Tp_row, Tm_row, mpT_row, mmT_row,
      w1T_row, b1_row, w2T_row, b2_row);
  // Col pass: xmid -> out (with residual)
  stu_pass_kernel<1><<<512, 256, 0, stream>>>(
      xmid, (float*)d_out, gamma_col, beta_col, Tp_col, Tm_col, mpT_col,
      mmT_col, w1T_col, b1_col, w2T_col, b2_col);
}
